// SemanticIds_49529562858369
// MI455X (gfx1250) — matrix operand order repaired and checked by reading, not compiled, not run on hardware
//
#include <hip/hip_runtime.h>

#define B_SZ     16384
#define D_IN_SZ  1024
#define CB       512
#define K_CODES  1024
#define L_LAYERS 4

typedef __attribute__((ext_vector_type(16))) __bf16 v16bf;
typedef __attribute__((ext_vector_type(8)))  float  v8f;

union BF16Frag { v16bf v; unsigned short u[16]; };

__device__ __forceinline__ unsigned short f2bf(float f) {
    union { float f; unsigned u; } x; x.f = f;
    // round-to-nearest-even truncation to bf16
    return (unsigned short)((x.u + 0x7FFFu + ((x.u >> 16) & 1u)) >> 16);
}

// ---------------------------------------------------------------------------
// Kernel 0: codebooks fp32 -> bf16, and half squared norms 0.5*||c||^2
// ---------------------------------------------------------------------------
__global__ void __launch_bounds__(256) prep_codebooks(
    const float* __restrict__ cb, unsigned short* __restrict__ cbb,
    float* __restrict__ hn) {
    __shared__ float red[256];
    int code = blockIdx.x;                      // 0 .. L*K-1
    const float* src = cb + (size_t)code * CB;
    unsigned short* dst = cbb + (size_t)code * CB;
    int t = threadIdx.x;
    float s = 0.f;
    for (int i = t; i < CB; i += 256) {
        float v = src[i];
        dst[i] = f2bf(v);
        s += v * v;
    }
    red[t] = s; __syncthreads();
    for (int o = 128; o; o >>= 1) {
        if (t < o) red[t] += red[t + o];
        __syncthreads();
    }
    if (t == 0) hn[code] = 0.5f * red[0];
}

// ---------------------------------------------------------------------------
// Kernel 1: encoder GEMM  R = X(16384x1024) @ W(1024x512) + b  (bf16 WMMA)
// Block tile 128x64, 8 waves, each wave 2x2 accumulators of 16x16, K step 32.
// ---------------------------------------------------------------------------
__global__ void __launch_bounds__(256) encoder_gemm(
    const float* __restrict__ X, const float* __restrict__ W,
    const float* __restrict__ bias, float* __restrict__ R) {
    __shared__ unsigned short As[128][34];   // row-major, padded (17-bank stride)
    __shared__ unsigned short Bs[64][34];    // transposed W tile: Bs[n][k]
    int m0 = blockIdx.x * 128;
    int n0 = blockIdx.y * 64;
    int t = threadIdx.x, lane = t & 31, w = t >> 5;
    int wm = (w & 3) * 32, wn = (w >> 2) * 32;
    int lq = lane & 15, half = lane >> 4;

    v8f acc[2][2] = {};

    for (int k0 = 0; k0 < D_IN_SZ; k0 += 32) {
        // stage A: 128 rows x 32 K (fp32 -> bf16), 1024 float4 loads
        #pragma unroll
        for (int j = 0; j < 4; ++j) {
            int idx = t + j * 256;               // 0..1023
            int mi = idx >> 3, kk = (idx & 7) * 4;
            const float4 f = *(const float4*)(X + (size_t)(m0 + mi) * D_IN_SZ + k0 + kk);
            As[mi][kk + 0] = f2bf(f.x); As[mi][kk + 1] = f2bf(f.y);
            As[mi][kk + 2] = f2bf(f.z); As[mi][kk + 3] = f2bf(f.w);
        }
        // stage B transposed: source 32 K-rows x 64 cols -> Bs[n][k]
        #pragma unroll
        for (int j = 0; j < 2; ++j) {
            int idx = t + j * 256;               // 0..511
            int kk = idx >> 4, nn = (idx & 15) * 4;
            const float4 f = *(const float4*)(W + (size_t)(k0 + kk) * CB + n0 + nn);
            Bs[nn + 0][kk] = f2bf(f.x); Bs[nn + 1][kk] = f2bf(f.y);
            Bs[nn + 2][kk] = f2bf(f.z); Bs[nn + 3][kk] = f2bf(f.w);
        }
        __syncthreads();

        BF16Frag a[2], b[2];
        #pragma unroll
        for (int s = 0; s < 2; ++s) {
            // A fragment: lane holds row (lane&15); K = khalf..khalf+15
            int row = wm + s * 16 + lq;
            int ka = half * 16;
            #pragma unroll
            for (int i = 0; i < 16; ++i) a[s].u[i] = As[row][ka + i];
            // B fragment: lane holds col; K = khalf..khalf+15
            int col = wn + s * 16 + lq;
            int kb = half * 16;
            #pragma unroll
            for (int i = 0; i < 16; ++i) b[s].u[i] = Bs[col][kb + i];
        }
        #pragma unroll
        for (int sm = 0; sm < 2; ++sm)
            #pragma unroll
            for (int sn = 0; sn < 2; ++sn)
                acc[sm][sn] = __builtin_amdgcn_wmma_f32_16x16x32_bf16(
                    false, a[sm].v, false, b[sn].v, (short)0, acc[sm][sn], false, false);
        __syncthreads();
    }

    // store C (+ bias): VGPR r <-> row r + 8*(lane>=16), col = lane&15
    #pragma unroll
    for (int sm = 0; sm < 2; ++sm)
        #pragma unroll
        for (int sn = 0; sn < 2; ++sn) {
            int col = n0 + wn + sn * 16 + lq;
            float bv = bias[col];
            #pragma unroll
            for (int r = 0; r < 8; ++r) {
                int row = m0 + wm + sm * 16 + half * 8 + r;
                R[(size_t)row * CB + col] = acc[sm][sn][r] + bv;
            }
        }
}

// ---------------------------------------------------------------------------
// Kernel 2 (x4): one RQ layer. Block owns 32 rows (2 row-tiles of 16), so
// every codebook B fragment loaded from L2 feeds TWO WMMAs (halves L2
// traffic vs 16-row blocks, doubles WMMA ILP behind each global load).
// score[m][k] = r_m . c_k - 0.5||c_k||^2 ; argmax = argmin distance.
// Then fp32 residual update in place.
// ---------------------------------------------------------------------------
__global__ void __launch_bounds__(256) rq_layer(
    float* __restrict__ R, const unsigned short* __restrict__ cbb,
    const float* __restrict__ cbf, const float* __restrict__ hn,
    int* __restrict__ ids, int layer) {
    __shared__ unsigned short rs[32][520];   // bf16 residual tile, padded
    __shared__ float wbv[8][32];
    __shared__ int   wbi[8][32];
    __shared__ int   bestIdx[32];

    int m0 = blockIdx.x * 32;
    int t = threadIdx.x, lane = t & 31, w = t >> 5;
    int lq = lane & 15, half = lane >> 4;

    // stage residual tile 32x512 fp32 -> bf16 LDS (4096 float4 loads)
    #pragma unroll
    for (int j = 0; j < 16; ++j) {
        int idx = t + j * 256;                 // float4 units
        int mi = idx >> 7, c4 = (idx & 127) * 4;
        const float4 f = *(const float4*)(R + (size_t)(m0 + mi) * CB + c4);
        rs[mi][c4 + 0] = f2bf(f.x); rs[mi][c4 + 1] = f2bf(f.y);
        rs[mi][c4 + 2] = f2bf(f.z); rs[mi][c4 + 3] = f2bf(f.w);
    }
    __syncthreads();

    // each wave scores 128 codes against 32 rows: 2x8 accumulators,
    // K loop hoists both A fragments, B fragment reused for both row tiles
    v8f acc[2][8] = {};
    for (int ks = 0; ks < CB; ks += 32) {
        BF16Frag a[2];
        int ka = half * 16;
        #pragma unroll
        for (int s = 0; s < 2; ++s) {
            int row = s * 16 + lq;
            #pragma unroll
            for (int i = 0; i < 16; ++i) a[s].u[i] = rs[row][ks + ka + i];
        }
        #pragma unroll
        for (int nt = 0; nt < 8; ++nt) {
            int code = w * 128 + nt * 16 + lq;
            const unsigned short* p = cbb + (size_t)code * CB + ks + half * 16;
            BF16Frag b;
            #pragma unroll
            for (int i = 0; i < 16; ++i) b.u[i] = p[i];   // 2x global_load_b128
            acc[0][nt] = __builtin_amdgcn_wmma_f32_16x16x32_bf16(
                false, a[0].v, false, b.v, (short)0, acc[0][nt], false, false);
            acc[1][nt] = __builtin_amdgcn_wmma_f32_16x16x32_bf16(
                false, a[1].v, false, b.v, (short)0, acc[1][nt], false, false);
        }
    }

    // per-lane running argmax (score = dot - halfnorm), 16 rows per lane
    float bv[2][8]; int bi[2][8];
    #pragma unroll
    for (int s = 0; s < 2; ++s)
        #pragma unroll
        for (int r = 0; r < 8; ++r) { bv[s][r] = -__builtin_inff(); bi[s][r] = 0x7FFFFFFF; }
    #pragma unroll
    for (int nt = 0; nt < 8; ++nt) {
        int code = w * 128 + nt * 16 + lq;
        float h = hn[code];
        #pragma unroll
        for (int s = 0; s < 2; ++s)
            #pragma unroll
            for (int r = 0; r < 8; ++r) {
                float v = acc[s][nt][r] - h;
                if (v > bv[s][r] || (v == bv[s][r] && code < bi[s][r])) {
                    bv[s][r] = v; bi[s][r] = code;
                }
            }
    }
    // butterfly reduce across the 16 lanes holding the same rows
    #pragma unroll
    for (int s = 0; s < 2; ++s)
        #pragma unroll
        for (int r = 0; r < 8; ++r)
            #pragma unroll
            for (int off = 8; off >= 1; off >>= 1) {
                float ov = __shfl_xor(bv[s][r], off, 32);
                int   oi = __shfl_xor(bi[s][r], off, 32);
                if (ov > bv[s][r] || (ov == bv[s][r] && oi < bi[s][r])) {
                    bv[s][r] = ov; bi[s][r] = oi;
                }
            }
    if (lq == 0) {
        #pragma unroll
        for (int s = 0; s < 2; ++s)
            #pragma unroll
            for (int r = 0; r < 8; ++r) {
                wbv[w][s * 16 + half * 8 + r] = bv[s][r];
                wbi[w][s * 16 + half * 8 + r] = bi[s][r];
            }
    }
    __syncthreads();

    // cross-wave reduce, emit ids
    if (t < 32) {
        float best = wbv[0][t]; int bidx = wbi[0][t];
        #pragma unroll
        for (int ww = 1; ww < 8; ++ww) {
            float v = wbv[ww][t]; int i2 = wbi[ww][t];
            if (v > best || (v == best && i2 < bidx)) { best = v; bidx = i2; }
        }
        bestIdx[t] = bidx;
        ids[(size_t)(m0 + t) * L_LAYERS + layer] = bidx;
    }
    __syncthreads();

    // fp32 residual update in place: r -= cb[best]
    #pragma unroll
    for (int j = 0; j < 16; ++j) {
        int idx = t + j * 256;
        int mi = idx >> 7, c4 = (idx & 127) * 4;
        size_t ro = (size_t)(m0 + mi) * CB + c4;
        const float4 cv = *(const float4*)(cbf + (size_t)bestIdx[mi] * CB + c4);
        float4 rv = *(const float4*)(R + ro);
        rv.x -= cv.x; rv.y -= cv.y; rv.z -= cv.z; rv.w -= cv.w;
        *(float4*)(R + ro) = rv;
    }
}

// ---------------------------------------------------------------------------
// Kernel 3: decoder (ids as float) @ dec_W + dec_b, plus int32 ids to out tail
// ---------------------------------------------------------------------------
__global__ void __launch_bounds__(256) decoder_kernel(
    const int* __restrict__ ids, const float* __restrict__ dW,
    const float* __restrict__ db, float* __restrict__ recon,
    int* __restrict__ outIds) {
    size_t g = (size_t)blockIdx.x * 256 + threadIdx.x;
    int b = (int)(g / CB), n = (int)(g % CB);
    float s = db[n];
    #pragma unroll
    for (int l = 0; l < L_LAYERS; ++l)
        s += (float)ids[b * L_LAYERS + l] * dW[l * CB + n];
    recon[g] = s;
    if (n < L_LAYERS) outIds[b * L_LAYERS + n] = ids[b * L_LAYERS + n];
}

// ---------------------------------------------------------------------------
extern "C" void kernel_launch(void* const* d_in, const int* in_sizes, int n_in,
                              void* d_out, int out_size, void* d_ws, size_t ws_size,
                              hipStream_t stream) {
    const float* X   = (const float*)d_in[0];   // [B, D_IN]
    const float* eW  = (const float*)d_in[1];   // [D_IN, CB]
    const float* eb  = (const float*)d_in[2];   // [CB]
    const float* cbf = (const float*)d_in[3];   // [L, K, CB]
    const float* dW  = (const float*)d_in[4];   // [L, CB]
    const float* db  = (const float*)d_in[5];   // [CB]

    char* ws = (char*)d_ws;
    float* R = (float*)ws;                       ws += (size_t)B_SZ * CB * sizeof(float);
    unsigned short* cbb = (unsigned short*)ws;   ws += (size_t)L_LAYERS * K_CODES * CB * sizeof(unsigned short);
    float* hn = (float*)ws;                      ws += (size_t)L_LAYERS * K_CODES * sizeof(float);
    int* ids = (int*)ws;

    float* recon = (float*)d_out;
    int* outIds = (int*)((float*)d_out + (size_t)B_SZ * CB);

    prep_codebooks<<<L_LAYERS * K_CODES, 256, 0, stream>>>(cbf, cbb, hn);
    encoder_gemm<<<dim3(B_SZ / 128, CB / 64), 256, 0, stream>>>(X, eW, eb, R);
    for (int l = 0; l < L_LAYERS; ++l)
        rq_layer<<<B_SZ / 32, 256, 0, stream>>>(
            R, cbb + (size_t)l * K_CODES * CB, cbf + (size_t)l * K_CODES * CB,
            hn + (size_t)l * K_CODES, ids, l);
    decoder_kernel<<<(B_SZ * CB) / 256, 256, 0, stream>>>(ids, dW, db, recon, outIds);
}
